// Model_83743272337575
// MI455X (gfx1250) — matrix operand order repaired and checked by reading, not compiled, not run on hardware
//
#include <hip/hip_runtime.h>
#include <hip/hip_bf16.h>
#include <math.h>

// ---------------- problem constants (match reference) ----------------
constexpr int Bc = 16, Sc = 512, Nc = 321, Dc = 512, DFc = 2048, Pc = 96;
constexpr int LAYERSc = 2, Kf = 3;
constexpr int NHEAD = 4;                 // M wavelet levels + 1
constexpr int BNc = Bc * Nc;             // 5136
constexpr int BHc = Bc * NHEAD;          // 64 attention batches
constexpr int NPAD = 352;                // head dim 321 padded to 11*32 (aligned fast path)
constexpr float ALPHAc = 0.3f;

typedef __attribute__((ext_vector_type(16))) __bf16 v16bf;
typedef __attribute__((ext_vector_type(8)))  float  v8f;

union WFrag { unsigned u[8]; v16bf v; };

__device__ __forceinline__ unsigned short f2bf(float f) {
  unsigned u = __builtin_bit_cast(unsigned, f);
  u += 0x7FFFu + ((u >> 16) & 1u);          // round-to-nearest-even
  return (unsigned short)(u >> 16);
}

#if __has_builtin(__builtin_amdgcn_cvt_pk_bf16_f32)
typedef __attribute__((ext_vector_type(2))) __bf16 v2bf;
__device__ __forceinline__ unsigned pack2bf(float lo, float hi) {
  v2bf r = __builtin_amdgcn_cvt_pk_bf16_f32(lo, hi);   // v_cvt_pk_bf16_f32
  return __builtin_bit_cast(unsigned, r);
}
#else
__device__ __forceinline__ unsigned pack2bf(float lo, float hi) {
  return (unsigned)f2bf(lo) | ((unsigned)f2bf(hi) << 16);
}
#endif

__device__ __forceinline__ float gelu_exact(float x) {
  return 0.5f * x * (1.0f + erff(x * 0.70710678118654752f));
}

// ---------------- WMMA GEMM: C[M,N] = A[M,K] x B (+bias, +act) ----------------
// BT=true : B stored as (N,K) row-major (ldb over K)  -> C += A * B^T
// BT=false: B stored as (K,N) row-major (ldb over N)  -> C += A * B
// ACT: 0 none, 1 exact GELU. Batched over blockIdx.z with element strides sA/sB/sC.
template <bool BT, int ACT>
__launch_bounds__(128)
__global__ void gemm_wmma(const float* __restrict__ A, const float* __restrict__ Bm,
                          const float* __restrict__ bias, float* __restrict__ C,
                          int M, int N, int K, int lda, int ldb, int ldc,
                          long long sA, long long sB, long long sC) {
  constexpr int LST = 40;                  // 80B row stride: 16B-aligned b128 LDS stores
  __shared__ unsigned short As[64 * LST];  // A tile: [row][k]   (64 x 32 bf16)
  __shared__ unsigned short Bs[64 * LST];  // B tile: [col][k]   (64 x 32 bf16)
  A  += (long long)blockIdx.z * sA;
  Bm += (long long)blockIdx.z * sB;
  C  += (long long)blockIdx.z * sC;

  const int tid   = threadIdx.x;
  const int wave  = tid >> 5;              // 4 wave32 waves
  const int lane  = tid & 31;
  const int mlane = lane & 15;
  const int hi    = lane >> 4;
  const int m0 = blockIdx.y * 64;
  const int n0 = blockIdx.x * 64;

  v8f acc[4];
  #pragma unroll
  for (int i = 0; i < 4; ++i)
    #pragma unroll
    for (int r = 0; r < 8; ++r) acc[i][r] = 0.0f;

  const bool ldaOK = ((lda & 3) == 0);
  const bool ldbOK = ((ldb & 3) == 0);
  const int kTiles = (K + 31) >> 5;
  for (int kt = 0; kt < kTiles; ++kt) {
    const int k0 = kt << 5;
    const bool kFull = (k0 + 32 <= K);

    // ---- stage A 64x32 (f32 -> bf16) ----
    if (kFull && (m0 + 64 <= M) && ldaOK) {
      // vector fast path: 4x global_load_b128 + 2x ds_store_b128 per thread
      int r = tid >> 1, hh = (tid & 1) << 4;
      const float* src = A + (size_t)(m0 + r) * lda + (k0 + hh);
      float4 a0 = *(const float4*)(src + 0);
      float4 a1 = *(const float4*)(src + 4);
      float4 a2 = *(const float4*)(src + 8);
      float4 a3 = *(const float4*)(src + 12);
      uint4 p0, p1;
      p0.x = pack2bf(a0.x, a0.y); p0.y = pack2bf(a0.z, a0.w);
      p0.z = pack2bf(a1.x, a1.y); p0.w = pack2bf(a1.z, a1.w);
      p1.x = pack2bf(a2.x, a2.y); p1.y = pack2bf(a2.z, a2.w);
      p1.z = pack2bf(a3.x, a3.y); p1.w = pack2bf(a3.z, a3.w);
      *(uint4*)&As[r * LST + hh] = p0;
      *(uint4*)&As[r * LST + hh + 8] = p1;
    } else {
      for (int i = tid; i < 64 * 32; i += 128) {
        int r = i >> 5, kk = i & 31;
        int gm = m0 + r, gk = k0 + kk;
        float v = (gm < M && gk < K) ? A[(size_t)gm * lda + gk] : 0.0f;
        As[r * LST + kk] = f2bf(v);
      }
    }

    // ---- stage B into Bs[col][k] ----
    if (BT) {
      if (kFull && (n0 + 64 <= N) && ldbOK) {
        int c = tid >> 1, hh = (tid & 1) << 4;
        const float* src = Bm + (size_t)(n0 + c) * ldb + (k0 + hh);
        float4 a0 = *(const float4*)(src + 0);
        float4 a1 = *(const float4*)(src + 4);
        float4 a2 = *(const float4*)(src + 8);
        float4 a3 = *(const float4*)(src + 12);
        uint4 p0, p1;
        p0.x = pack2bf(a0.x, a0.y); p0.y = pack2bf(a0.z, a0.w);
        p0.z = pack2bf(a1.x, a1.y); p0.w = pack2bf(a1.z, a1.w);
        p1.x = pack2bf(a2.x, a2.y); p1.y = pack2bf(a2.z, a2.w);
        p1.z = pack2bf(a3.x, a3.y); p1.w = pack2bf(a3.z, a3.w);
        *(uint4*)&Bs[c * LST + hh] = p0;
        *(uint4*)&Bs[c * LST + hh + 8] = p1;
      } else {
        for (int i = tid; i < 64 * 32; i += 128) {
          int c = i >> 5, kk = i & 31;
          int gn = n0 + c, gk = k0 + kk;
          float v = (gn < N && gk < K) ? Bm[(size_t)gn * ldb + gk] : 0.0f;
          Bs[c * LST + kk] = f2bf(v);
        }
      }
    } else {
      if (kFull && (n0 + 64 <= N) && ldbOK) {
        // thread handles a k-pair x 8 columns; transpose via 8 packed ds_store_b32
        int kp = (tid >> 3) << 1;          // 0,2,...,30
        int nb = (tid & 7) << 3;           // 0,8,...,56
        const float* s0 = Bm + (size_t)(k0 + kp) * ldb + (n0 + nb);
        const float* s1 = s0 + ldb;
        float4 r00 = *(const float4*)(s0);
        float4 r01 = *(const float4*)(s0 + 4);
        float4 r10 = *(const float4*)(s1);
        float4 r11 = *(const float4*)(s1 + 4);
        *(unsigned*)&Bs[(nb + 0) * LST + kp] = pack2bf(r00.x, r10.x);
        *(unsigned*)&Bs[(nb + 1) * LST + kp] = pack2bf(r00.y, r10.y);
        *(unsigned*)&Bs[(nb + 2) * LST + kp] = pack2bf(r00.z, r10.z);
        *(unsigned*)&Bs[(nb + 3) * LST + kp] = pack2bf(r00.w, r10.w);
        *(unsigned*)&Bs[(nb + 4) * LST + kp] = pack2bf(r01.x, r11.x);
        *(unsigned*)&Bs[(nb + 5) * LST + kp] = pack2bf(r01.y, r11.y);
        *(unsigned*)&Bs[(nb + 6) * LST + kp] = pack2bf(r01.z, r11.z);
        *(unsigned*)&Bs[(nb + 7) * LST + kp] = pack2bf(r01.w, r11.w);
      } else {
        for (int i = tid; i < 64 * 32; i += 128) {
          int kk = i >> 6, c = i & 63;
          int gn = n0 + c, gk = k0 + kk;
          float v = (gn < N && gk < K) ? Bm[(size_t)gk * ldb + gn] : 0.0f;
          Bs[c * LST + kk] = f2bf(v);
        }
      }
    }
    __syncthreads();

    // prefetch next A tile into cache hierarchy (global_prefetch_b8)
    if (kt + 1 < kTiles) {
      int pr = m0 + (tid >> 1);
      if (pr < M) __builtin_prefetch(&A[(size_t)pr * lda + (k0 + 32)], 0, 1);
    }

    // A fragment: CDNA5 16-bit A 16x32 layout (ISA 7.12.2)
    WFrag af;
    const unsigned short* ap = &As[(wave * 16 + mlane) * LST];
    #pragma unroll
    for (int v2 = 0; v2 < 8; ++v2) {
      int kk = ((v2 < 4) ? (2 * v2) : (16 + 2 * (v2 - 4))) + 8 * hi;
      af.u[v2] = *(const unsigned*)(ap + kk);
    }
    #pragma unroll
    for (int ns = 0; ns < 4; ++ns) {
      // B fragment: same K order as A (VGPR j<4 -> K=2j+8h, j>=4 -> K=16+2(j-4)+8h)
      WFrag bfm;
      const unsigned short* bp = &Bs[(ns * 16 + mlane) * LST];
      #pragma unroll
      for (int v2 = 0; v2 < 8; ++v2) {
        int kk = ((v2 < 4) ? (2 * v2) : (16 + 2 * (v2 - 4))) + 8 * hi;
        bfm.u[v2] = *(const unsigned*)(bp + kk);
      }
      acc[ns] = __builtin_amdgcn_wmma_f32_16x16x32_bf16(
          false, af.v, false, bfm.v, (short)0, acc[ns], false, false);
    }
    __syncthreads();
  }
  // epilogue: C/D layout VGPR r -> M=r (lanes 0-15) / M=8+r (lanes 16-31), N=lane&15
  #pragma unroll
  for (int ns = 0; ns < 4; ++ns) {
    int gn = n0 + ns * 16 + mlane;
    if (gn >= N) continue;
    float bv = bias ? bias[gn] : 0.0f;
    #pragma unroll
    for (int r = 0; r < 8; ++r) {
      int gm = m0 + wave * 16 + r + 8 * hi;
      if (gm < M) {
        float v = acc[ns][r] + bv;
        if (ACT == 1) v = gelu_exact(v);
        C[(size_t)gm * ldc + gn] = v;
      }
    }
  }
}

// ---------------- per-(b,n) mean / stdev over S ----------------
__global__ void stats_kernel(const float* __restrict__ x, float* __restrict__ mean,
                             float* __restrict__ stdev) {
  int bn = blockIdx.x;
  int b = bn / Nc, n = bn % Nc;
  const float* p = x + (size_t)b * Sc * Nc + n;
  int tid = threadIdx.x;
  float s = 0.f, s2 = 0.f;
  for (int i = tid; i < Sc; i += 256) {
    float v = p[(size_t)i * Nc];
    s += v; s2 += v * v;
  }
  __shared__ float r1[256], r2[256];
  r1[tid] = s; r2[tid] = s2;
  __syncthreads();
  for (int st = 128; st > 0; st >>= 1) {
    if (tid < st) { r1[tid] += r1[tid + st]; r2[tid] += r2[tid + st]; }
    __syncthreads();
  }
  if (tid == 0) {
    float mu = r1[0] / (float)Sc;
    float var = r2[0] / (float)Sc - mu * mu;
    mean[bn] = mu;
    stdev[bn] = sqrtf(var + 1e-5f);
  }
}

// xT[bn, s] = (x[b,s,n]-mean)/std   (normalized + transposed for the embedding GEMM)
__global__ void normT_kernel(const float* __restrict__ x, const float* __restrict__ mean,
                             const float* __restrict__ stdev, float* __restrict__ xT) {
  size_t i = (size_t)blockIdx.x * 256 + threadIdx.x;
  if (i >= (size_t)BNc * Sc) return;
  int s = (int)(i & (Sc - 1));
  size_t bn = i >> 9;
  int b = (int)(bn / Nc), n = (int)(bn % Nc);
  xT[i] = (x[((size_t)b * Sc + s) * Nc + n] - mean[bn]) / stdev[bn];
}

// ---------------- SWT decomposition (3 levels, circular, dilated, K=3) ----------------
__global__ void swt_dec_kernel(const float* __restrict__ hin, const float* __restrict__ h0l,
                               const float* __restrict__ h1l, float* __restrict__ coeffs) {
  int bn = blockIdx.x;
  int n = bn % Nc;
  int tid = threadIdx.x;
  __shared__ float cur[Dc];
  for (int d = tid; d < Dc; d += 256) cur[d] = hin[(size_t)bn * Dc + d];
  float f0[3], f1[3];
  #pragma unroll
  for (int t = 0; t < 3; ++t) { f0[t] = h0l[n * 3 + t]; f1[t] = h1l[n * 3 + t]; }
  __syncthreads();
  int dil = 1;
  float na[2];
  for (int lvl = 0; lvl < 3; ++lvl) {
    int padL = 2 * dil - (3 * dil) / 2;   // pad_total - pad_r from reference
    int j = 0;
    for (int d = tid; d < Dc; d += 256, ++j) {
      float a = 0.f, dt = 0.f;
      #pragma unroll
      for (int t = 0; t < 3; ++t) {
        int idx = (d + t * dil - padL + Dc) & (Dc - 1);
        float c = cur[idx];
        a += c * f0[t];
        dt += c * f1[t];
      }
      na[j] = a;
      coeffs[((size_t)bn * NHEAD + (3 - lvl)) * Dc + d] = dt;
    }
    __syncthreads();
    j = 0;
    for (int d = tid; d < Dc; d += 256, ++j) cur[d] = na[j];
    __syncthreads();
    dil <<= 1;
  }
  for (int d = tid; d < Dc; d += 256) coeffs[(size_t)bn * NHEAD * Dc + d] = cur[d];
}

// ---------------- SWT reconstruction ----------------
__global__ void swt_rec_kernel(const float* __restrict__ c, const float* __restrict__ g0l,
                               const float* __restrict__ g1l, float* __restrict__ outp) {
  int bn = blockIdx.x;
  int n = bn % Nc;
  int tid = threadIdx.x;
  __shared__ float cur[Dc];
  __shared__ float det[Dc];
  float f0[3], f1[3];
  #pragma unroll
  for (int t = 0; t < 3; ++t) { f0[t] = g0l[n * 3 + t]; f1[t] = g1l[n * 3 + t]; }
  for (int d = tid; d < Dc; d += 256) cur[d] = c[(size_t)bn * NHEAD * Dc + d];
  int dil = 4;
  float nv[2];
  for (int i = 0; i < 3; ++i) {
    for (int d = tid; d < Dc; d += 256)
      det[d] = c[((size_t)bn * NHEAD + 1 + i) * Dc + d];
    __syncthreads();
    int padL = (3 * dil) / 2;
    int j = 0;
    for (int d = tid; d < Dc; d += 256, ++j) {
      float a = 0.f;
      #pragma unroll
      for (int t = 0; t < 3; ++t) {
        int idx = (d + t * dil - padL + Dc) & (Dc - 1);
        a += cur[idx] * f0[t] + det[idx] * f1[t];
      }
      nv[j] = 0.5f * a;
    }
    __syncthreads();
    j = 0;
    for (int d = tid; d < Dc; d += 256, ++j) cur[d] = nv[j];
    __syncthreads();
    dil >>= 1;
  }
  for (int d = tid; d < Dc; d += 256) outp[(size_t)bn * Dc + d] = cur[d];
}

// ---------------- permutes ----------------
// src[((b*Nc+n)*4+m)*Dc+e] -> dst[((b*4+m)*Dc+e)*NPAD+n], zero-filled pad
__global__ void perm_qkv_pad_kernel(const float* __restrict__ src, float* __restrict__ dst) {
  size_t i = (size_t)blockIdx.x * 256 + threadIdx.x;
  size_t total = (size_t)BHc * Dc * NPAD;
  if (i >= total) return;
  int n = (int)(i % NPAD);
  size_t t = i / NPAD;
  int e = (int)(t & (Dc - 1));
  t >>= 9;
  int m = (int)(t & 3);
  int b = (int)(t >> 2);
  dst[i] = (n < Nc) ? src[(((size_t)b * Nc + n) * NHEAD + m) * Dc + e] : 0.0f;
}

// src[((b*4+h)*Dc+l)*Nc+hd] -> dst[((b*Nc+hd)*4+h)*Dc+l]
__global__ void perm_attnout_kernel(const float* __restrict__ src, float* __restrict__ dst) {
  size_t i = (size_t)blockIdx.x * 256 + threadIdx.x;
  size_t total = (size_t)Bc * Nc * NHEAD * Dc;
  if (i >= total) return;
  int l = (int)(i & (Dc - 1));
  size_t t = i >> 9;
  int hh = (int)(t & 3);
  t >>= 2;
  int hd = (int)(t % Nc);
  int b = (int)(t / Nc);
  dst[i] = src[(((size_t)b * NHEAD + hh) * Dc + l) * Nc + hd];
}

// ---------------- row sum of squares (q/k norms, len=321, padded rows) ----------------
__global__ void rowsumsq_kernel(const float* __restrict__ X, float* __restrict__ outp,
                                int len, int rowStride) {
  int row = blockIdx.x;
  const float* p = X + (size_t)row * rowStride;
  int tid = threadIdx.x;
  float s = 0.f;
  for (int i = tid; i < len; i += 128) { float v = p[i]; s += v * v; }
  __shared__ float red[128];
  red[tid] = s;
  __syncthreads();
  for (int st = 64; st > 0; st >>= 1) {
    if (tid < st) red[tid] += red[tid + st];
    __syncthreads();
  }
  if (tid == 0) outp[row] = red[0];
}

// ---------------- geometric-attention wedge + softmax (row-wise, in-place) ----------------
__global__ void attn_softmax_kernel(float* __restrict__ dot, const float* __restrict__ qn2,
                                    const float* __restrict__ kn2) {
  int row = blockIdx.x, tid = threadIdx.x;
  int bh = row >> 9;
  float q2 = qn2[row];
  const float* k2 = kn2 + (size_t)bh * Dc;
  float* p = dot + (size_t)row * Dc;
  __shared__ float srow[Dc];
  __shared__ float red[256];
  const float scale = 1.0f / sqrtf(321.0f);
  float mx = -3.4e38f;
  for (int s = tid; s < Dc; s += 256) {
    float d = p[s];
    float w = sqrtf(fmaxf(q2 * k2[s] - d * d, 0.0f) + 1e-8f);
    float sc = ((1.0f - ALPHAc) * d + ALPHAc * w) * scale;
    srow[s] = sc;
    mx = fmaxf(mx, sc);
  }
  red[tid] = mx;
  __syncthreads();
  for (int st = 128; st > 0; st >>= 1) {
    if (tid < st) red[tid] = fmaxf(red[tid], red[tid + st]);
    __syncthreads();
  }
  mx = red[0];
  __syncthreads();
  float ls = 0.f;
  for (int s = tid; s < Dc; s += 256) {
    float e = __expf(srow[s] - mx);
    srow[s] = e;
    ls += e;
  }
  red[tid] = ls;
  __syncthreads();
  for (int st = 128; st > 0; st >>= 1) {
    if (tid < st) red[tid] += red[tid + st];
    __syncthreads();
  }
  float inv = 1.0f / red[0];
  for (int s = tid; s < Dc; s += 256) p[s] = srow[s] * inv;
}

// ---------------- LayerNorm (dim=512, optional fused residual) ----------------
__global__ void ln_kernel(const float* __restrict__ x, const float* __restrict__ res,
                          const float* __restrict__ g, const float* __restrict__ b,
                          float* __restrict__ outp) {
  int row = blockIdx.x, tid = threadIdx.x;
  const size_t base = (size_t)row * Dc;
  float v0 = x[base + tid], v1 = x[base + tid + 256];
  if (res) { v0 += res[base + tid]; v1 += res[base + tid + 256]; }
  __shared__ float red[256];
  red[tid] = v0 + v1;
  __syncthreads();
  for (int st = 128; st > 0; st >>= 1) {
    if (tid < st) red[tid] += red[tid + st];
    __syncthreads();
  }
  float mu = red[0] / (float)Dc;
  __syncthreads();
  float d0 = v0 - mu, d1 = v1 - mu;
  red[tid] = d0 * d0 + d1 * d1;
  __syncthreads();
  for (int st = 128; st > 0; st >>= 1) {
    if (tid < st) red[tid] += red[tid + st];
    __syncthreads();
  }
  float rstd = rsqrtf(red[0] / (float)Dc + 1e-5f);
  outp[base + tid] = d0 * rstd * g[tid] + b[tid];
  outp[base + tid + 256] = d1 * rstd * g[tid + 256] + b[tid + 256];
}

// ---------------- final: dec[b,p,n] = pdec[b,n,p]*stdev + mean ----------------
__global__ void final_out_kernel(const float* __restrict__ pdec, const float* __restrict__ mean,
                                 const float* __restrict__ stdev, float* __restrict__ outp) {
  size_t i = (size_t)blockIdx.x * 256 + threadIdx.x;
  size_t total = (size_t)Bc * Pc * Nc;
  if (i >= total) return;
  int n = (int)(i % Nc);
  size_t t = i / Nc;
  int p = (int)(t % Pc);
  int b = (int)(t / Pc);
  size_t bn = (size_t)b * Nc + n;
  outp[i] = pdec[bn * Pc + p] * stdev[bn] + mean[bn];
}

__global__ void copy_ms_kernel(const float* __restrict__ mean, const float* __restrict__ stdev,
                               float* __restrict__ outp) {
  int i = blockIdx.x * 256 + threadIdx.x;
  if (i >= BNc) return;
  outp[i] = mean[i];
  outp[BNc + i] = stdev[i];
}

// ---------------- host orchestration ----------------
extern "C" void kernel_launch(void* const* d_in, const int* in_sizes, int n_in,
                              void* d_out, int out_size, void* d_ws, size_t ws_size,
                              hipStream_t stream) {
  (void)in_sizes; (void)n_in; (void)out_size; (void)ws_size;
  const float* x_enc = (const float*)d_in[0];
  const float* emb_W = (const float*)d_in[1];
  const float* emb_b = (const float*)d_in[2];
  const float* h0 = (const float*)d_in[3];
  const float* h1 = (const float*)d_in[4];
  const float* g0 = (const float*)d_in[5];
  const float* g1 = (const float*)d_in[6];
  const float* Wq = (const float*)d_in[7];
  const float* bq = (const float*)d_in[8];
  const float* Wk = (const float*)d_in[9];
  const float* bk = (const float*)d_in[10];
  const float* Wv = (const float*)d_in[11];
  const float* bv = (const float*)d_in[12];
  const float* Wo = (const float*)d_in[13];
  const float* bo = (const float*)d_in[14];
  const float* W1 = (const float*)d_in[15];
  const float* b1 = (const float*)d_in[16];
  const float* W2 = (const float*)d_in[17];
  const float* b2 = (const float*)d_in[18];
  const float* ln1_g = (const float*)d_in[19];
  const float* ln1_b = (const float*)d_in[20];
  const float* ln2_g = (const float*)d_in[21];
  const float* ln2_b = (const float*)d_in[22];
  const float* lnf_g = (const float*)d_in[23];
  const float* lnf_b = (const float*)d_in[24];
  const float* proj_W = (const float*)d_in[25];
  const float* proj_b = (const float*)d_in[26];
  float* out = (float*)d_out;

  // ---- workspace carve (floats) ----
  size_t off = 0;
  float* base = (float*)d_ws;
  auto carve = [&](size_t nElems) { float* p = base + off; off += nElems; return p; };
  const size_t bigC = (size_t)BNc * NHEAD * Dc;    // 10,518,528
  const size_t bigP = (size_t)BHc * Dc * NPAD;     // padded q/k/v: 11,534,336
  float* means  = carve(BNc);
  float* stdev  = carve(BNc);
  float* h      = carve((size_t)BNc * Dc);
  float* x1     = carve((size_t)BNc * Dc);
  float* t1     = carve((size_t)BNc * Dc);      // swt_rec output
  float* t2     = carve((size_t)BNc * Dc);      // sublayer output
  float* coeffs = carve(bigC);                  // also rec_in (reused)
  float* qb     = carve(bigC);                  // also xT, also attention O (reused)
  float* kb     = carve(bigC);
  float* vb     = carve(bigC);
  float* qT     = carve(bigP);
  float* kT     = carve(bigP);
  float* vT     = carve(bigP);
  float* dotb   = carve((size_t)BHc * Dc * Dc); // 16.8M
  float* yb     = carve((size_t)BNc * DFc);
  float* qn2    = carve((size_t)BHc * Dc);
  float* kn2    = carve((size_t)BHc * Dc);
  float* pdec   = carve((size_t)BNc * Pc);

  // ---- stats + normalized transpose ----
  stats_kernel<<<BNc, 256, 0, stream>>>(x_enc, means, stdev);
  {
    size_t total = (size_t)BNc * Sc;
    normT_kernel<<<(unsigned)((total + 255) / 256), 256, 0, stream>>>(x_enc, means, stdev, qb);
  }
  // h[bn,d] = xT[bn,:] . emb_W[d,:] + emb_b   (BT GEMM, K=S)
  gemm_wmma<true, 0><<<dim3(Dc / 64, (BNc + 63) / 64, 1), 128, 0, stream>>>(
      qb, emb_W, emb_b, h, BNc, Dc, Sc, Sc, Sc, Dc, 0, 0, 0);

  const unsigned gpPad = (unsigned)(((size_t)BHc * Dc * NPAD + 255) / 256);
  const unsigned gpO   = (unsigned)(((size_t)Bc * Nc * NHEAD * Dc + 255) / 256);

  for (int l = 0; l < LAYERSc; ++l) {
    const float* h0l = h0 + (size_t)l * Nc * Kf;
    const float* h1l = h1 + (size_t)l * Nc * Kf;
    const float* g0l = g0 + (size_t)l * Nc * Kf;
    const float* g1l = g1 + (size_t)l * Nc * Kf;

    swt_dec_kernel<<<BNc, 256, 0, stream>>>(h, h0l, h1l, coeffs);

    const int Mqkv = BNc * NHEAD;  // 20544
    gemm_wmma<true, 0><<<dim3(Dc / 64, Mqkv / 64, 1), 128, 0, stream>>>(
        coeffs, Wq + (size_t)l * Dc * Dc, bq + (size_t)l * Dc, qb,
        Mqkv, Dc, Dc, Dc, Dc, Dc, 0, 0, 0);
    gemm_wmma<true, 0><<<dim3(Dc / 64, Mqkv / 64, 1), 128, 0, stream>>>(
        coeffs, Wk + (size_t)l * Dc * Dc, bk + (size_t)l * Dc, kb,
        Mqkv, Dc, Dc, Dc, Dc, Dc, 0, 0, 0);
    gemm_wmma<true, 0><<<dim3(Dc / 64, Mqkv / 64, 1), 128, 0, stream>>>(
        coeffs, Wv + (size_t)l * Dc * Dc, bv + (size_t)l * Dc, vb,
        Mqkv, Dc, Dc, Dc, Dc, Dc, 0, 0, 0);

    perm_qkv_pad_kernel<<<gpPad, 256, 0, stream>>>(qb, qT);
    perm_qkv_pad_kernel<<<gpPad, 256, 0, stream>>>(kb, kT);
    perm_qkv_pad_kernel<<<gpPad, 256, 0, stream>>>(vb, vT);

    rowsumsq_kernel<<<BHc * Dc, 128, 0, stream>>>(qT, qn2, Nc, NPAD);
    rowsumsq_kernel<<<BHc * Dc, 128, 0, stream>>>(kT, kn2, Nc, NPAD);

    // dot[bh] = qT[bh] (512xNPAD) . kT[bh]^T  (zero pads contribute nothing)
    gemm_wmma<true, 0><<<dim3(Dc / 64, Dc / 64, BHc), 128, 0, stream>>>(
        qT, kT, nullptr, dotb, Dc, Dc, NPAD, NPAD, NPAD, Dc,
        (long long)Dc * NPAD, (long long)Dc * NPAD, (long long)Dc * Dc);

    attn_softmax_kernel<<<BHc * Dc, 256, 0, stream>>>(dotb, qn2, kn2);

    // O[bh] (512x321) = P[bh] (512x512) . vT[bh] (512xNPAD rows)  -> reuse qb as O
    gemm_wmma<false, 0><<<dim3((Nc + 63) / 64, Dc / 64, BHc), 128, 0, stream>>>(
        dotb, vT, nullptr, qb, Dc, Nc, Dc, Dc, NPAD, Nc,
        (long long)Dc * Dc, (long long)Dc * NPAD, (long long)Dc * Nc);

    perm_attnout_kernel<<<gpO, 256, 0, stream>>>(qb, coeffs);  // rec_in reuses coeffs
    swt_rec_kernel<<<BNc, 256, 0, stream>>>(coeffs, g0l, g1l, t1);

    gemm_wmma<true, 0><<<dim3(Dc / 64, (BNc + 63) / 64, 1), 128, 0, stream>>>(
        t1, Wo + (size_t)l * Dc * Dc, bo + (size_t)l * Dc, t2,
        BNc, Dc, Dc, Dc, Dc, Dc, 0, 0, 0);

    ln_kernel<<<BNc, 256, 0, stream>>>(h, t2, ln1_g + (size_t)l * Dc, ln1_b + (size_t)l * Dc, x1);

    gemm_wmma<true, 1><<<dim3(DFc / 64, (BNc + 63) / 64, 1), 128, 0, stream>>>(
        x1, W1 + (size_t)l * DFc * Dc, b1 + (size_t)l * DFc, yb,
        BNc, DFc, Dc, Dc, Dc, DFc, 0, 0, 0);
    gemm_wmma<true, 0><<<dim3(Dc / 64, (BNc + 63) / 64, 1), 128, 0, stream>>>(
        yb, W2 + (size_t)l * Dc * DFc, b2 + (size_t)l * Dc, t2,
        BNc, Dc, DFc, DFc, DFc, Dc, 0, 0, 0);

    ln_kernel<<<BNc, 256, 0, stream>>>(x1, t2, ln2_g + (size_t)l * Dc, ln2_b + (size_t)l * Dc, h);
  }

  ln_kernel<<<BNc, 256, 0, stream>>>(h, nullptr, lnf_g, lnf_b, x1);
  gemm_wmma<true, 0><<<dim3((Pc + 63) / 64, (BNc + 63) / 64, 1), 128, 0, stream>>>(
      x1, proj_W, proj_b, pdec, BNc, Pc, Dc, Dc, Dc, Pc, 0, 0, 0);
  {
    size_t total = (size_t)Bc * Pc * Nc;
    final_out_kernel<<<(unsigned)((total + 255) / 256), 256, 0, stream>>>(pdec, means, stdev, out);
  }
  copy_ms_kernel<<<(BNc + 255) / 256, 256, 0, stream>>>(means, stdev, out + (size_t)Bc * Pc * Nc);
}
